// ParticleEnergyPredictor_V2_81836306858289
// MI455X (gfx1250) — compile-verified, matrix operand order checked by reading, not run on hardware
//
#include <hip/hip_runtime.h>
#include <hip/hip_bf16.h>
#include <math.h>

// ---------------- problem constants ----------------
#define NB_     250
#define BATCH_  512
#define M_ROWS  (BATCH_ * NB_)   // 128000 rows, multiple of 16
#define F_PAD   32               // 15 features padded to 32 (one WMMA K step)
#define EH_     250
#define EHP     256              // energy hidden padded
#define PH_     64               // prior hidden
#define EPSF    1e-8f
#define RCUT    4.8f
#define BN_EPSF 1e-5f

typedef _Float16 v16h __attribute__((ext_vector_type(16)));
typedef float    v8f  __attribute__((ext_vector_type(8)));
typedef unsigned v4u  __attribute__((ext_vector_type(4)));

union AFrag { v16h v; unsigned u[8]; v4u q[2]; _Float16 h[16]; };

#define CDIV(a,b) (((a)+(b)-1)/(b))

// ---------------- utility ----------------
__global__ void k_zero(float* p, int n) {
    int i = blockIdx.x * blockDim.x + threadIdx.x;
    if (i < n) p[i] = 0.f;
}

// Pack f32 weight (sR x sC), optionally transposed, into padded f16 WMMA
// B-fragment order for a logical K x N matrix:
//   dst[ ((ntile*(K/32)+kblock)*32 + lane)*16 + 2*v + p ] = Wlog[k][n]
// so each lane's fragment is 32 contiguous bytes (two b128 loads).
__global__ void k_packB(const float* __restrict__ src, _Float16* __restrict__ dst,
                        int sR, int sC, int K, int N, int tr) {
    int i = blockIdx.x * blockDim.x + threadIdx.x;
    if (i >= K * N) return;
    int k = i / N, n = i - k * N;
    float val = 0.f;
    if (tr) { if (n < sR && k < sC) val = src[n * sC + k]; }   // Wlog = src^T
    else    { if (k < sR && n < sC) val = src[k * sC + n]; }
    int ntile = n >> 4, l16 = n & 15;
    int kblock = k >> 5, k32 = k & 31;
    int g = k32 >> 4, rem = k32 & 15;
    int hi = rem >> 3, rem8 = rem & 7;
    int vv = rem8 >> 1, p = rem8 & 1;
    int v = (g << 2) | vv;
    int lane = (hi << 4) | l16;
    size_t idx = ((size_t)(ntile * (K >> 5) + kblock) * 32 + lane) * 16 + 2 * v + p;
    dst[idx] = (_Float16)val;
}

// ---------------- features ----------------
__global__ void k_features(const float* __restrict__ dr_, const float* __restrict__ O_,
                           const float* __restrict__ nO_,
                           _Float16* __restrict__ fh, _Float16* __restrict__ fqh,
                           float* __restrict__ Rr, float* __restrict__ drn_) {
    int r = blockIdx.x * blockDim.x + threadIdx.x;
    if (r >= M_ROWS) return;
    int b = r / NB_;
    float d0 = dr_[r*3+0] + EPSF, d1 = dr_[r*3+1] + EPSF, d2 = dr_[r*3+2] + EPSF;
    float R = sqrtf(d0*d0 + d1*d1 + d2*d2);
    float inv = 1.f / R;
    float n0 = d0*inv, n1 = d1*inv, n2 = d2*inv;
    Rr[r] = R; drn_[r*3+0]=n0; drn_[r*3+1]=n1; drn_[r*3+2]=n2;
    const float* O  = O_  + (size_t)b*9;
    const float* nO = nO_ + (size_t)r*9;
    float f[15];
    #pragma unroll
    for (int k = 0; k < 3; ++k) {
        f[k]   = n0*O[0*3+k] + n1*O[1*3+k] + n2*O[2*3+k];
        f[3+k] = n0*nO[0*3+k] + n1*nO[1*3+k] + n2*nO[2*3+k];
    }
    #pragma unroll
    for (int k = 0; k < 3; ++k)
        #pragma unroll
        for (int l = 0; l < 3; ++l)
            f[6 + k*3 + l] = O[0*3+k]*nO[0*3+l] + O[1*3+k]*nO[1*3+l] + O[2*3+k]*nO[2*3+l];
    #pragma unroll
    for (int j = 0; j < F_PAD; ++j) {
        float v = (j < 15) ? f[j] : 0.f;
        fh [(size_t)r*F_PAD + j] = (_Float16)v;
        fqh[(size_t)r*F_PAD + j] = (_Float16)(v*v);
    }
}

// ---------------- WMMA GEMM: out = [tanh](A[MxK] @ Wlog[KxN] + bias) ----------------
// A f16 row-major; Wf pre-packed B-fragment order. One wave -> 16 x (NT*16) tile.
// K compile-time => fully unrolled (K/32)*NT WMMAs per wave.
template<int K, int NT>
__global__ void k_gemm(const _Float16* __restrict__ A, const _Float16* __restrict__ Wf,
                       const float* __restrict__ bias, int biasLen,
                       _Float16* __restrict__ out, int M, int N, int act) {
    const int lane = threadIdx.x & 31;
    const int wid  = blockIdx.x * (blockDim.x >> 5) + (threadIdx.x >> 5);
    const int groups_n = (N >> 4) / NT;
    const int groups   = (M >> 4) * groups_n;
    if (wid >= groups) return;                 // wave-uniform: EXEC stays all-ones
    const int mtile = wid / groups_n, ng = wid - mtile * groups_n;
    const int l16 = lane & 15, hi = lane >> 4;
    const int mrow = (mtile << 4) + l16;
    const size_t arow = (size_t)mrow * K;
    constexpr int KB = K >> 5;
    v8f zero = {};
    v8f acc[NT];
    #pragma unroll
    for (int t = 0; t < NT; ++t) acc[t] = zero;
    #pragma unroll
    for (int kb = 0; kb < KB; ++kb) {
        if (kb + 1 < KB)
            __builtin_prefetch((const void*)(A + arow + (kb + 1) * 32), 0, 1);
        AFrag a;
        a.q[0] = *(const v4u*)(A + arow + kb * 32 + hi * 8);        // 8 halves, 16B aligned
        a.q[1] = *(const v4u*)(A + arow + kb * 32 + 16 + hi * 8);
        #pragma unroll
        for (int t = 0; t < NT; ++t) {
            const int ntile = ng * NT + t;
            const v16h bf = *(const v16h*)(Wf + ((size_t)(ntile * KB + kb) * 32 + lane) * 16);
            acc[t] = __builtin_amdgcn_wmma_f32_16x16x32_f16(false, a.v, false, bf,
                                                            (short)0, acc[t], false, false);
        }
    }
    #pragma unroll
    for (int t = 0; t < NT; ++t) {
        int ncol = (ng * NT + t) * 16 + l16;
        float bv = (bias && ncol < biasLen) ? bias[ncol] : 0.f;
        #pragma unroll
        for (int v = 0; v < 8; ++v) {
            int row = (mtile << 4) + v + (hi << 3);
            float x = acc[t][v] + bv;
            if (act) x = tanhf(x);
            out[(size_t)row * N + ncol] = (_Float16)x;
        }
    }
}

// ---------------- BatchNorm (channel = neighbor index n = r % NB_) ----------------
__global__ void k_bnstats(const _Float16* __restrict__ z, float* __restrict__ mean,
                          float* __restrict__ rstd, int Hvalid, int Npad) {
    const int c = blockIdx.x;
    __shared__ float s1[256], s2[256];
    float a = 0.f, b = 0.f;
    const int total = BATCH_ * Hvalid;
    for (int i = threadIdx.x; i < total; i += blockDim.x) {
        int bb = i / Hvalid, h = i - bb * Hvalid;
        float v = (float)z[((size_t)bb * NB_ + c) * Npad + h];
        a += v; b += v * v;
    }
    s1[threadIdx.x] = a; s2[threadIdx.x] = b; __syncthreads();
    for (int s = 128; s > 0; s >>= 1) {
        if ((int)threadIdx.x < s) { s1[threadIdx.x] += s1[threadIdx.x+s]; s2[threadIdx.x] += s2[threadIdx.x+s]; }
        __syncthreads();
    }
    if (threadIdx.x == 0) {
        float m = s1[0] / total;
        float var = s2[0] / total - m * m;
        mean[c] = m; rstd[c] = rsqrtf(var + BN_EPSF);
    }
}

__global__ void k_bntanh(const _Float16* __restrict__ z, _Float16* __restrict__ actout,
                         const float* __restrict__ mean, const float* __restrict__ rstd,
                         const float* __restrict__ g, const float* __restrict__ bt,
                         int Hvalid, int Npad) {
    int i = blockIdx.x * blockDim.x + threadIdx.x;
    if (i >= M_ROWS * Npad) return;
    int r = i / Npad, h = i - r * Npad, c = r % NB_;
    float v = 0.f;
    if (h < Hvalid) {
        float xh = ((float)z[i] - mean[c]) * rstd[c];
        v = tanhf(g[c] * xh + bt[c]);
    }
    actout[i] = (_Float16)v;
}

__global__ void k_bnbwd_reduce(const _Float16* __restrict__ z, const _Float16* __restrict__ act,
                               const _Float16* __restrict__ gmat, const float* __restrict__ gvec,
                               const float* __restrict__ rowscale, float gscale,
                               const float* __restrict__ mean, const float* __restrict__ rstd,
                               _Float16* __restrict__ dy_out, float* __restrict__ S1,
                               float* __restrict__ S2, int Hvalid, int Npad) {
    int r = blockIdx.x, c = r % NB_;
    float rs = rowscale ? rowscale[r] : 1.f;
    __shared__ float s1[256], s2[256];
    float a = 0.f, b = 0.f;
    for (int h = threadIdx.x; h < Npad; h += blockDim.x) {
        size_t i = (size_t)r * Npad + h;
        float dy = 0.f;
        if (h < Hvalid) {
            float g = gscale * rs;
            if (gmat) g *= (float)gmat[i];
            if (gvec) g *= gvec[h];
            float av = (float)act[i];
            dy = g * (1.f - av * av);
            float xh = ((float)z[i] - mean[c]) * rstd[c];
            a += dy; b += dy * xh;
        }
        dy_out[i] = (_Float16)dy;
    }
    s1[threadIdx.x] = a; s2[threadIdx.x] = b; __syncthreads();
    for (int s = 128; s > 0; s >>= 1) {
        if ((int)threadIdx.x < s) { s1[threadIdx.x] += s1[threadIdx.x+s]; s2[threadIdx.x] += s2[threadIdx.x+s]; }
        __syncthreads();
    }
    if (threadIdx.x == 0) { atomicAdd(&S1[c], s1[0]); atomicAdd(&S2[c], s2[0]); }
}

__global__ void k_bnbwd_apply(const _Float16* __restrict__ z, const _Float16* __restrict__ dy,
                              _Float16* __restrict__ dz, const float* __restrict__ mean,
                              const float* __restrict__ rstd, const float* __restrict__ g,
                              const float* __restrict__ S1, const float* __restrict__ S2,
                              float invcnt, int Hvalid, int Npad) {
    int i = blockIdx.x * blockDim.x + threadIdx.x;
    if (i >= M_ROWS * Npad) return;
    int r = i / Npad, h = i - r * Npad, c = r % NB_;
    float v = 0.f;
    if (h < Hvalid) {
        float xh = ((float)z[i] - mean[c]) * rstd[c];
        v = g[c] * rstd[c] * ((float)dy[i] - S1[c] * invcnt - xh * S2[c] * invcnt);
    }
    dz[i] = (_Float16)v;
}

__global__ void k_tanh_bwd(const _Float16* __restrict__ gmat, const _Float16* __restrict__ act,
                           _Float16* __restrict__ dz, int n) {
    int i = blockIdx.x * blockDim.x + threadIdx.x;
    if (i >= n) return;
    float a = (float)act[i];
    dz[i] = (_Float16)((float)gmat[i] * (1.f - a * a));
}

// ---------------- heads ----------------
__global__ void k_final_energy(const _Float16* __restrict__ act2, const float* __restrict__ W3,
                               const float* __restrict__ b3, float* __restrict__ Ssym) {
    int r = blockIdx.x * blockDim.x + threadIdx.x;
    if (r >= M_ROWS) return;
    float s = b3[0];
    for (int h = 0; h < EH_; ++h) s += (float)act2[(size_t)r * EHP + h] * W3[h];
    atomicAdd(Ssym, s);
}

__global__ void k_final_prior(const _Float16* __restrict__ p1, const float* __restrict__ W2,
                              const float* __restrict__ b2, const float* __restrict__ Rr,
                              float* __restrict__ pe_prior, float* __restrict__ enc,
                              float* __restrict__ ex) {
    int r = blockIdx.x * blockDim.x + threadIdx.x;
    if (r >= M_ROWS) return;
    int b = r / NB_;
    float s = b2[0];
    for (int h = 0; h < PH_; ++h) s += (float)p1[(size_t)r * PH_ + h] * W2[h];
    float e = tanhf(s);
    float rc = Rr[r] / RCUT;
    float x = expf(-powf(rc, 12.f));
    enc[r] = e; ex[r] = x;
    atomicAdd(&pe_prior[b], x * e);     // PRE_FACTOR = 1
}

__global__ void k_pe(const float* __restrict__ Ssym, const float* __restrict__ pe_prior,
                     float* __restrict__ out_pe) {
    int b = blockIdx.x * blockDim.x + threadIdx.x;
    if (b < BATCH_) out_pe[b] = Ssym[0] + pe_prior[b];
}

__global__ void k_prior_seed(const float* __restrict__ enc, const float* __restrict__ ex,
                             float* __restrict__ rowscale) {
    int r = blockIdx.x * blockDim.x + threadIdx.x;
    if (r >= M_ROWS) return;
    float e = enc[r];
    rowscale[r] = ex[r] * (1.f - e * e);
}

// ---------------- geometry backward ----------------
__global__ void k_bwd_geom(const _Float16* __restrict__ gfe, const _Float16* __restrict__ gfq,
                           const _Float16* __restrict__ fh, const float* __restrict__ drn,
                           const float* __restrict__ Rr, const float* __restrict__ O_,
                           const float* __restrict__ nO_, const float* __restrict__ enc,
                           const float* __restrict__ ex, float* __restrict__ force,
                           float* __restrict__ gO) {
    int r = blockIdx.x * blockDim.x + threadIdx.x;
    if (r >= M_ROWS) return;
    int b = r / NB_;
    float gf[15];
    #pragma unroll
    for (int j = 0; j < 15; ++j) {
        size_t i = (size_t)r * F_PAD + j;
        gf[j] = (float)gfe[i] + 2.f * (float)fh[i] * (float)gfq[i];
    }
    const float* O  = O_  + (size_t)b*9;
    const float* nO = nO_ + (size_t)r*9;
    float dn[3] = { drn[r*3], drn[r*3+1], drn[r*3+2] };
    float R = Rr[r];
    float rc = R / RCUT;
    float gR = ex[r] * enc[r] * (-12.f * powf(rc, 11.f) / RCUT);
    float gdn[3];
    #pragma unroll
    for (int i = 0; i < 3; ++i) {
        float s = 0.f;
        #pragma unroll
        for (int k = 0; k < 3; ++k)
            s += gf[k] * O[i*3+k] + gf[3+k] * nO[i*3+k];
        gdn[i] = s;
    }
    float dot = gdn[0]*dn[0] + gdn[1]*dn[1] + gdn[2]*dn[2];
    float invR = 1.f / R;
    #pragma unroll
    for (int i = 0; i < 3; ++i) {
        float gdr = (gdn[i] - dot * dn[i]) * invR + gR * dn[i];
        atomicAdd(&force[b*3 + i], gdr);
    }
    #pragma unroll
    for (int i = 0; i < 3; ++i)
        #pragma unroll
        for (int k = 0; k < 3; ++k) {
            float v = gf[k] * dn[i];
            #pragma unroll
            for (int l = 0; l < 3; ++l) v += gf[6 + k*3 + l] * nO[i*3 + l];
            atomicAdd(&gO[(size_t)b*9 + i*3 + k], v);
        }
}

__global__ void k_torque(const float* __restrict__ gO, const float* __restrict__ O_,
                         float* __restrict__ out_t) {
    int b = blockIdx.x * blockDim.x + threadIdx.x;
    if (b >= BATCH_) return;
    const float* g = gO + (size_t)b*9;
    const float* O = O_ + (size_t)b*9;
    float t0 = 0.f, t1 = 0.f, t2 = 0.f;
    #pragma unroll
    for (int k = 0; k < 3; ++k) {
        float u0 = g[0*3+k], u1 = g[1*3+k], u2 = g[2*3+k];
        float v0 = O[0*3+k], v1 = O[1*3+k], v2 = O[2*3+k];
        t0 += u1*v2 - u2*v1;
        t1 += u2*v0 - u0*v2;
        t2 += u0*v1 - u1*v0;
    }
    out_t[b*3+0] = t0; out_t[b*3+1] = t1; out_t[b*3+2] = t2;
}

// ---------------- host-side GEMM dispatch ----------------
template<int K, int NT>
static void launch_gemm(const _Float16* A, const _Float16* Wf, const float* bias, int blen,
                        _Float16* o, int N, int act, hipStream_t s) {
    int groups = (M_ROWS >> 4) * ((N >> 4) / NT);
    k_gemm<K, NT><<<CDIV(groups, 8), 256, 0, s>>>(A, Wf, bias, blen, o, M_ROWS, N, act);
}

// ---------------- launch ----------------
extern "C" void kernel_launch(void* const* d_in, const int* in_sizes, int n_in,
                              void* d_out, int out_size, void* d_ws, size_t ws_size,
                              hipStream_t stream) {
    const float* dr    = (const float*)d_in[0];
    const float* Omat  = (const float*)d_in[1];
    const float* nO    = (const float*)d_in[2];
    const float* pW0   = (const float*)d_in[3];
    const float* pb0   = (const float*)d_in[4];
    const float* pW1   = (const float*)d_in[5];
    const float* pb1   = (const float*)d_in[6];
    const float* pg1   = (const float*)d_in[7];
    const float* pbt1  = (const float*)d_in[8];
    const float* pW2   = (const float*)d_in[9];
    const float* pb2   = (const float*)d_in[10];
    const float* eW0   = (const float*)d_in[11];
    const float* eb0   = (const float*)d_in[12];
    const float* eW1   = (const float*)d_in[13];
    const float* eb1   = (const float*)d_in[14];
    const float* eg1   = (const float*)d_in[15];
    const float* ebt1  = (const float*)d_in[16];
    const float* eW2   = (const float*)d_in[17];
    const float* eb2   = (const float*)d_in[18];
    const float* eg2   = (const float*)d_in[19];
    const float* ebt2  = (const float*)d_in[20];
    const float* eW3   = (const float*)d_in[21];
    const float* eb3   = (const float*)d_in[22];
    float* out = (float*)d_out;       // [force 1536 | torque 1536 | pe 512]

    const size_t M = M_ROWS;
    char* base = (char*)d_ws;
    auto alloc = [&](size_t bytes) -> void* {
        void* p = (void*)base;
        base += (bytes + 255) & ~(size_t)255;
        return p;
    };
    // activations (f16)
    _Float16* fh   = (_Float16*)alloc(M * F_PAD * 2);
    _Float16* fqh  = (_Float16*)alloc(M * F_PAD * 2);
    _Float16* e0   = (_Float16*)alloc(M * EHP * 2);
    _Float16* z1   = (_Float16*)alloc(M * EHP * 2);
    _Float16* e1   = (_Float16*)alloc(M * EHP * 2);
    _Float16* z2   = (_Float16*)alloc(M * EHP * 2);
    _Float16* e2   = (_Float16*)alloc(M * EHP * 2);
    _Float16* p0   = (_Float16*)alloc(M * PH_ * 2);
    _Float16* zp1  = (_Float16*)alloc(M * PH_ * 2);
    _Float16* p1   = (_Float16*)alloc(M * PH_ * 2);
    // grads (f16)
    _Float16* dyE  = (_Float16*)alloc(M * EHP * 2);
    _Float16* dzE  = (_Float16*)alloc(M * EHP * 2);
    _Float16* gE   = (_Float16*)alloc(M * EHP * 2);
    _Float16* dyP  = (_Float16*)alloc(M * PH_ * 2);
    _Float16* dzP  = (_Float16*)alloc(M * PH_ * 2);
    _Float16* gP   = (_Float16*)alloc(M * PH_ * 2);
    _Float16* gfe  = (_Float16*)alloc(M * F_PAD * 2);
    _Float16* gfq  = (_Float16*)alloc(M * F_PAD * 2);
    // packed weights in B-fragment order (f16)
    _Float16* eW0f = (_Float16*)alloc(F_PAD * EHP * 2);
    _Float16* eW1f = (_Float16*)alloc(EHP * EHP * 2);
    _Float16* eW2f = (_Float16*)alloc(EHP * EHP * 2);
    _Float16* pW0f = (_Float16*)alloc(F_PAD * PH_ * 2);
    _Float16* pW1f = (_Float16*)alloc(PH_ * PH_ * 2);
    _Float16* eW0t = (_Float16*)alloc(EHP * F_PAD * 2);
    _Float16* eW1t = (_Float16*)alloc(EHP * EHP * 2);
    _Float16* eW2t = (_Float16*)alloc(EHP * EHP * 2);
    _Float16* pW0t = (_Float16*)alloc(PH_ * F_PAD * 2);
    _Float16* pW1t = (_Float16*)alloc(PH_ * PH_ * 2);
    // f32 scratch
    float* Rr       = (float*)alloc(M * 4);
    float* drn      = (float*)alloc(M * 3 * 4);
    float* enc      = (float*)alloc(M * 4);
    float* exf      = (float*)alloc(M * 4);
    float* rowscale = (float*)alloc(M * 4);
    float* meanE1   = (float*)alloc(NB_ * 4);
    float* rstdE1   = (float*)alloc(NB_ * 4);
    float* meanE2   = (float*)alloc(NB_ * 4);
    float* rstdE2   = (float*)alloc(NB_ * 4);
    float* meanP    = (float*)alloc(NB_ * 4);
    float* rstdP    = (float*)alloc(NB_ * 4);
    float* S1       = (float*)alloc(NB_ * 4);
    float* S2       = (float*)alloc(NB_ * 4);
    float* Ssym     = (float*)alloc(4);
    float* pe_prior = (float*)alloc(BATCH_ * 4);
    float* gO       = (float*)alloc(BATCH_ * 9 * 4);

    const int T = 256;
    // zero accumulators + force region of d_out
    k_zero<<<CDIV(BATCH_*3, T), T, 0, stream>>>(out, BATCH_*3);
    k_zero<<<1, T, 0, stream>>>(Ssym, 1);
    k_zero<<<CDIV(BATCH_, T), T, 0, stream>>>(pe_prior, BATCH_);
    k_zero<<<CDIV(BATCH_*9, T), T, 0, stream>>>(gO, BATCH_*9);

    // pack weights into WMMA B-fragment layouts (forward + transposed)
    k_packB<<<CDIV(F_PAD*EHP, T), T, 0, stream>>>(eW0, eW0f, 15, EH_, F_PAD, EHP, 0);
    k_packB<<<CDIV(EHP*EHP, T), T, 0, stream>>>(eW1, eW1f, EH_, EH_, EHP, EHP, 0);
    k_packB<<<CDIV(EHP*EHP, T), T, 0, stream>>>(eW2, eW2f, EH_, EH_, EHP, EHP, 0);
    k_packB<<<CDIV(F_PAD*PH_, T), T, 0, stream>>>(pW0, pW0f, 15, PH_, F_PAD, PH_, 0);
    k_packB<<<CDIV(PH_*PH_, T), T, 0, stream>>>(pW1, pW1f, PH_, PH_, PH_, PH_, 0);
    k_packB<<<CDIV(EHP*F_PAD, T), T, 0, stream>>>(eW0, eW0t, 15, EH_, EHP, F_PAD, 1);
    k_packB<<<CDIV(EHP*EHP, T), T, 0, stream>>>(eW1, eW1t, EH_, EH_, EHP, EHP, 1);
    k_packB<<<CDIV(EHP*EHP, T), T, 0, stream>>>(eW2, eW2t, EH_, EH_, EHP, EHP, 1);
    k_packB<<<CDIV(PH_*F_PAD, T), T, 0, stream>>>(pW0, pW0t, 15, PH_, PH_, F_PAD, 1);
    k_packB<<<CDIV(PH_*PH_, T), T, 0, stream>>>(pW1, pW1t, PH_, PH_, PH_, PH_, 1);

    // features
    k_features<<<CDIV(M_ROWS, T), T, 0, stream>>>(dr, Omat, nO, fh, fqh, Rr, drn);

    // ---- energy net forward ----
    launch_gemm<F_PAD, 4>(fh, eW0f, eb0, EH_, e0, EHP, 1, stream);   // e0 = tanh(feats@W0+b0)
    launch_gemm<EHP, 4>(e0, eW1f, eb1, EH_, z1, EHP, 0, stream);     // z1
    k_bnstats<<<NB_, T, 0, stream>>>(z1, meanE1, rstdE1, EH_, EHP);
    k_bntanh<<<CDIV(M_ROWS*EHP, T), T, 0, stream>>>(z1, e1, meanE1, rstdE1, eg1, ebt1, EH_, EHP);
    launch_gemm<EHP, 4>(e1, eW2f, eb2, EH_, z2, EHP, 0, stream);     // z2
    k_bnstats<<<NB_, T, 0, stream>>>(z2, meanE2, rstdE2, EH_, EHP);
    k_bntanh<<<CDIV(M_ROWS*EHP, T), T, 0, stream>>>(z2, e2, meanE2, rstdE2, eg2, ebt2, EH_, EHP);
    k_final_energy<<<CDIV(M_ROWS, T), T, 0, stream>>>(e2, eW3, eb3, Ssym);

    // ---- prior net forward ----
    launch_gemm<F_PAD, 4>(fqh, pW0f, pb0, PH_, p0, PH_, 1, stream);
    launch_gemm<PH_, 4>(p0, pW1f, pb1, PH_, zp1, PH_, 0, stream);
    k_bnstats<<<NB_, T, 0, stream>>>(zp1, meanP, rstdP, PH_, PH_);
    k_bntanh<<<CDIV(M_ROWS*PH_, T), T, 0, stream>>>(zp1, p1, meanP, rstdP, pg1, pbt1, PH_, PH_);
    k_final_prior<<<CDIV(M_ROWS, T), T, 0, stream>>>(p1, pW2, pb2, Rr, pe_prior, enc, exf);
    k_pe<<<CDIV(BATCH_, T), T, 0, stream>>>(Ssym, pe_prior, out + BATCH_*6);

    const float invE = 1.f / (float)(BATCH_ * EH_);
    const float invP = 1.f / (float)(BATCH_ * PH_);

    // ---- energy net backward ----
    // seed: d(sum pe)/d sym element = BATCH_ (scalar sum(sym) broadcast over batch)
    k_zero<<<1, T, 0, stream>>>(S1, NB_);
    k_zero<<<1, T, 0, stream>>>(S2, NB_);
    k_bnbwd_reduce<<<M_ROWS, T, 0, stream>>>(z2, e2, nullptr, eW3, nullptr, (float)BATCH_,
                                             meanE2, rstdE2, dyE, S1, S2, EH_, EHP);
    k_bnbwd_apply<<<CDIV(M_ROWS*EHP, T), T, 0, stream>>>(z2, dyE, dzE, meanE2, rstdE2, eg2,
                                                         S1, S2, invE, EH_, EHP);
    launch_gemm<EHP, 4>(dzE, eW2t, nullptr, 0, gE, EHP, 0, stream);  // g_e1 = dz2 @ W2^T
    k_zero<<<1, T, 0, stream>>>(S1, NB_);
    k_zero<<<1, T, 0, stream>>>(S2, NB_);
    k_bnbwd_reduce<<<M_ROWS, T, 0, stream>>>(z1, e1, gE, nullptr, nullptr, 1.f,
                                             meanE1, rstdE1, dyE, S1, S2, EH_, EHP);
    k_bnbwd_apply<<<CDIV(M_ROWS*EHP, T), T, 0, stream>>>(z1, dyE, dzE, meanE1, rstdE1, eg1,
                                                         S1, S2, invE, EH_, EHP);
    launch_gemm<EHP, 4>(dzE, eW1t, nullptr, 0, gE, EHP, 0, stream);  // g_e0 = dz1 @ W1^T
    k_tanh_bwd<<<CDIV(M_ROWS*EHP, T), T, 0, stream>>>(gE, e0, dzE, M_ROWS*EHP);
    launch_gemm<EHP, 2>(dzE, eW0t, nullptr, 0, gfe, F_PAD, 0, stream); // g_feats(energy)

    // ---- prior net backward ----
    k_prior_seed<<<CDIV(M_ROWS, T), T, 0, stream>>>(enc, exf, rowscale);
    k_zero<<<1, T, 0, stream>>>(S1, NB_);
    k_zero<<<1, T, 0, stream>>>(S2, NB_);
    k_bnbwd_reduce<<<M_ROWS, T, 0, stream>>>(zp1, p1, nullptr, pW2, rowscale, 1.f,
                                             meanP, rstdP, dyP, S1, S2, PH_, PH_);
    k_bnbwd_apply<<<CDIV(M_ROWS*PH_, T), T, 0, stream>>>(zp1, dyP, dzP, meanP, rstdP, pg1,
                                                         S1, S2, invP, PH_, PH_);
    launch_gemm<PH_, 4>(dzP, pW1t, nullptr, 0, gP, PH_, 0, stream);  // g_p0 = dzp1 @ pW1^T
    k_tanh_bwd<<<CDIV(M_ROWS*PH_, T), T, 0, stream>>>(gP, p0, dzP, M_ROWS*PH_);
    launch_gemm<PH_, 2>(dzP, pW0t, nullptr, 0, gfq, F_PAD, 0, stream); // g_featsq

    // ---- geometry backward: force + dE/dO, then torque ----
    k_bwd_geom<<<CDIV(M_ROWS, T), T, 0, stream>>>(gfe, gfq, fh, drn, Rr, Omat, nO,
                                                  enc, exf, out, gO);
    k_torque<<<CDIV(BATCH_, T), T, 0, stream>>>(gO, Omat, out + BATCH_*3);

    (void)in_sizes; (void)n_in; (void)out_size; (void)ws_size;
}
